// ComposerAttn_44753559224512
// MI455X (gfx1250) — matrix operand order repaired and checked by reading, not compiled, not run on hardware
//
#include <hip/hip_runtime.h>
#include <hip/hip_bf16.h>

// ---------------------------------------------------------------------------
// ComposerAttn for MI455X (gfx1250, wave32, WMMA).
//   N=16384 rows, NC=32 children, D=256, H=4 heads, DK=64.
// Strategy: bf16 v_wmma_f32_16x16x32_bf16 for all GEMMs; k-projection folded
// into q (scores = kv_in . (Wk^T q)) so only the v half needs the big GEMM.
// All weight B-matrices pre-converted to bf16 once (L2-resident) so GEMM
// inner loops are pure global_load_b128 / ds_load_b128 -> v_wmma.
// ---------------------------------------------------------------------------

typedef __attribute__((ext_vector_type(16))) __bf16 v16bf;
typedef __attribute__((ext_vector_type(8)))  __bf16 v8bf;
typedef __attribute__((ext_vector_type(8)))  float  v8f;

#define DIM     256
#define NCH     32
#define NH      4
#define RB      4                 // parent rows per block in attn kernel
#define MROWS   (RB * NCH)        // 128 (M dimension of the batched v-GEMM)
#define SA_STR  264               // padded bf16 row stride (16B aligned, bank-spread)
#define QSCALE  0.125f            // DK^-0.5 = 64^-0.5

// D = A*B + C, bf16 inputs, f32 accumulate (wave32, 16x16x32)
__device__ __forceinline__ v8f wmma_bf16(v16bf a, v16bf b, v8f c) {
  return __builtin_amdgcn_wmma_f32_16x16x32_bf16(false, a, false, b,
                                                 (short)0, c, false, false);
}

// A-fragment from a row-major f32 row pointer (already offset for this lane's
// K half): halves 0..15 = K sel+0..15.
__device__ __forceinline__ v16bf cvt_a(const float* p) {
  v16bf r;
#pragma unroll
  for (int i = 0; i < 8; ++i) { r[i] = (__bf16)p[i]; r[i + 8] = (__bf16)p[i + 8]; }
  return r;
}

// ---------------------------------------------------------------------------
// Kernel 0: convert weights to bf16 workspace (one pass, L2-resident after).
//   wkvv  <- Wkv rows 256..511 (v half), wqb <- Wq, woutb <- Wout
// ---------------------------------------------------------------------------
__global__ void __launch_bounds__(256)
cvt_weights(const float* __restrict__ Wkv, const float* __restrict__ Wq,
            const float* __restrict__ Wout, __bf16* __restrict__ wkvv,
            __bf16* __restrict__ wqb, __bf16* __restrict__ woutb) {
  int i = blockIdx.x * 256 + threadIdx.x;        // 65536 elements each
  wkvv[i]  = (__bf16)Wkv[(size_t)DIM * DIM + i];
  wqb[i]   = (__bf16)Wq[i];
  woutb[i] = (__bf16)Wout[i];
}

// ---------------------------------------------------------------------------
// Shared GEMM core: C[16 x 256] = A(16x256, f32 rows cvt->bf16) @ Wb^T (bf16).
// Wave-level: A hoisted in 8 frags; per nt, 8 k-step WMMAs with direct
// bf16 B loads (two global_load_b128 per frag). acc handed to epilogue.
// ---------------------------------------------------------------------------
__device__ __forceinline__ void load_a_frags(const float* arow, int half,
                                             v16bf a[8]) {
#pragma unroll
  for (int kb = 0; kb < 8; ++kb) a[kb] = cvt_a(arow + kb * 32 + half * 16);
}

__device__ __forceinline__ v8f gemm_tile(const v16bf a[8], const __bf16* wb,
                                         int nt, int col, int half) {
  v8f acc = {0, 0, 0, 0, 0, 0, 0, 0};
  const __bf16* brow = wb + (size_t)(nt * 16 + col) * DIM + half * 16;
  if (nt < 15)  // prefetch next B tile row (lowers to global_prefetch_b8)
    __builtin_prefetch(wb + (size_t)(nt * 16 + 16 + col) * DIM, 0, 1);
#pragma unroll
  for (int kb = 0; kb < 8; ++kb)
    acc = wmma_bf16(a[kb], *(const v16bf*)(brow + kb * 32), acc);
  return acc;
}

// ---------------------------------------------------------------------------
// Kernel 1: q = (parent @ Wq^T) * QSCALE  ->  q_out [N, 256] f32
// Block = 256 threads (8 waves); wave owns one 16-row M tile, loops 16 N tiles.
// ---------------------------------------------------------------------------
__global__ void __launch_bounds__(256)
qproj_wmma(const float* __restrict__ parent, const __bf16* __restrict__ wqb,
           float* __restrict__ q_out) {
  const int wave = threadIdx.x >> 5, lane = threadIdx.x & 31;
  const int half = (lane >> 4) & 1;
  const int col  = lane & 15;
  const int m0   = blockIdx.x * 128 + wave * 16;

  v16bf a[8];
  load_a_frags(parent + (size_t)(m0 + col) * DIM, half, a);

  for (int nt = 0; nt < 16; ++nt) {
    v8f acc = gemm_tile(a, wqb, nt, col, half);
    const int rbase = m0 + half * 8;
#pragma unroll
    for (int i = 0; i < 8; ++i)
      q_out[(size_t)(rbase + i) * DIM + nt * 16 + col] = acc[i] * QSCALE;
  }
}

// ---------------------------------------------------------------------------
// Kernel 2: per 4-row block — kv_in build, folded scores, softmax,
//           WMMA v-projection (M=128, K=256, N=256), fused ctx reduction.
// ---------------------------------------------------------------------------
__global__ void __launch_bounds__(256)
attn_wmma(const float* __restrict__ child_vecs,
          const long long* __restrict__ child_idx,
          const float* __restrict__ pos_emb,
          const float* __restrict__ Wkv,          // f32, k-half used here
          const float* __restrict__ q_ws,         // pre-scaled q
          const __bf16* __restrict__ wkvv,        // bf16 v-half [256][256]
          float* __restrict__ ctx_ws) {
  __shared__ __bf16 sA[MROWS * SA_STR];           // kv_in bf16 (A tiles)  ~67.6 KB
  __shared__ float  sKQ[RB][NH * DIM];            // folded k-proj of q     16 KB
  __shared__ float  sQ[RB][DIM];                  //                         4 KB
  __shared__ float  sAtt[RB][NCH][NH];            // scores -> att           2 KB
  __shared__ float  sCtx[RB][DIM];                //                         4 KB
  __shared__ int    sIdx[MROWS];

  const int tid = threadIdx.x;
  const int b0  = blockIdx.x * RB;

  // ---- phase 1: gather indices + q rows ----
  if (tid < MROWS) sIdx[tid] = (int)child_idx[(size_t)b0 * NCH + tid];
#pragma unroll
  for (int t = 0; t < RB; ++t) {
    int u = tid + t * 256;                        // u = r*256 + d
    sQ[u >> 8][u & 255] = q_ws[(size_t)b0 * DIM + u];
  }
  __syncthreads();

  // ---- phase 2a: kv_in = child + pos_emb[idx], convert to bf16 A tiles ----
  for (int u = tid; u < MROWS * DIM; u += 256) {
    int m = u >> 8, d = u & 255;
    float v = child_vecs[(size_t)(b0 * NCH + m) * DIM + d] +
              pos_emb[(size_t)sIdx[m] * DIM + d];
    sA[m * SA_STR + d] = (__bf16)v;
  }

  // ---- phase 2b: kq[r][h*256+d] = sum_dk Wkv[h*64+dk][d] * q[r][h*64+dk] ----
  for (int u = tid; u < NH * DIM; u += 256) {
    int h = u >> 8, d = u & 255;
    float acc[RB] = {0.f, 0.f, 0.f, 0.f};
#pragma unroll 4
    for (int dk = 0; dk < 64; ++dk) {
      float w = Wkv[(size_t)(h * 64 + dk) * DIM + d];
#pragma unroll
      for (int r = 0; r < RB; ++r) acc[r] += w * sQ[r][h * 64 + dk];
    }
#pragma unroll
    for (int r = 0; r < RB; ++r) sKQ[r][u] = acc[r];
  }
  __syncthreads();

  // ---- phase 3: scores[r][n][h] = kv_in[m] . kq[r][h] ----
  for (int c = tid; c < RB * NCH * NH; c += 256) {
    int r = c >> 7, n = (c >> 2) & 31, h = c & 3;
    const __bf16* ap = &sA[(r * NCH + n) * SA_STR];
    const float*  kq = &sKQ[r][h * DIM];
    float acc = 0.f;
#pragma unroll 8
    for (int d = 0; d < DIM; ++d) acc += (float)ap[d] * kq[d];
    sAtt[r][n][h] = acc;
  }
  __syncthreads();

  // ---- phase 4: softmax over children ----
  if (tid < RB * NH) {
    int r = tid >> 2, h = tid & 3;
    float mx = -1e30f;
    for (int n = 0; n < NCH; ++n) mx = fmaxf(mx, sAtt[r][n][h]);
    float s = 0.f;
    for (int n = 0; n < NCH; ++n) {
      float e = __expf(sAtt[r][n][h] - mx);
      sAtt[r][n][h] = e; s += e;
    }
    float inv = 1.0f / s;
    for (int n = 0; n < NCH; ++n) sAtt[r][n][h] *= inv;
  }
  __syncthreads();

  // ---- phase 5: WMMA v-projection + attention-weighted reduction ----
  const int wave = tid >> 5, lane = tid & 31;
  const int half = (lane >> 4) & 1;
  const v8f vzero = {0, 0, 0, 0, 0, 0, 0, 0};
  for (int j = 0; j < 2; ++j) {
    const int nt = wave * 2 + j;                  // 16 N-tiles over 8 waves
    const int h  = nt >> 2;                       // head (uniform per tile)
    v8f acc[8];
#pragma unroll
    for (int mt = 0; mt < 8; ++mt) acc[mt] = vzero;

    const __bf16* bbase = wkvv + (size_t)(nt * 16 + (lane & 15)) * DIM + half * 16;
    for (int kb = 0; kb < 8; ++kb) {
      v16bf bf = *(const v16bf*)(bbase + kb * 32);   // 32B contiguous per lane
#pragma unroll
      for (int mt = 0; mt < 8; ++mt) {
        int off = (mt * 16 + (lane & 15)) * SA_STR + kb * 32 + half * 16;
        v8bf lo = *(const v8bf*)(&sA[off]);          // ds_load_b128
        v8bf hi = *(const v8bf*)(&sA[off + 8]);      // ds_load_b128
        v16bf af = __builtin_shufflevector(lo, hi, 0, 1, 2, 3, 4, 5, 6, 7,
                                           8, 9, 10, 11, 12, 13, 14, 15);
        acc[mt] = wmma_bf16(af, bf, acc[mt]);
      }
    }
    // ctx[r][col] = sum_n att[r][n][h] * v[m][col]; lane holds fixed col.
    float part[RB] = {0.f, 0.f, 0.f, 0.f};
#pragma unroll
    for (int mt = 0; mt < 8; ++mt)
#pragma unroll
      for (int i = 0; i < 8; ++i) {
        int m = mt * 16 + half * 8 + i;           // global child row
        part[m >> 5] += sAtt[m >> 5][m & 31][h] * acc[mt][i];
      }
#pragma unroll
    for (int r = 0; r < RB; ++r) part[r] += __shfl_xor(part[r], 16, 32);
    if (lane < 16) {
#pragma unroll
      for (int r = 0; r < RB; ++r) sCtx[r][nt * 16 + lane] = part[r];
    }
  }
  __syncthreads();

#pragma unroll
  for (int t = 0; t < RB; ++t) {
    int u = tid + t * 256;
    ctx_ws[(size_t)b0 * DIM + u] = sCtx[u >> 8][u & 255];
  }
}

// ---------------------------------------------------------------------------
// Kernel 3: out = ctx @ Wout^T + bout; x = parent + out; LayerNorm(x).
// ---------------------------------------------------------------------------
__global__ void __launch_bounds__(256)
outproj_ln(const float* __restrict__ ctx_ws, const __bf16* __restrict__ woutb,
           const float* __restrict__ bout, const float* __restrict__ parent,
           const float* __restrict__ gamma, const float* __restrict__ beta,
           float* __restrict__ out) {
  __shared__ float sBuf[8][16][DIM];              // 128 KB: per-wave C tiles
  const int wave = threadIdx.x >> 5, lane = threadIdx.x & 31;
  const int half = (lane >> 4) & 1;
  const int col  = lane & 15;
  const int m0   = blockIdx.x * 128 + wave * 16;

  v16bf a[8];
  load_a_frags(ctx_ws + (size_t)(m0 + col) * DIM, half, a);

  for (int nt = 0; nt < 16; ++nt) {
    v8f acc = gemm_tile(a, woutb, nt, col, half);
#pragma unroll
    for (int i = 0; i < 8; ++i)
      sBuf[wave][half * 8 + i][nt * 16 + col] = acc[i];
  }
  __syncthreads();

  // LayerNorm: wave handles its 16 rows, lanes split the 256 columns.
  for (int rr = 0; rr < 16; ++rr) {
    const int g = m0 + rr;
    float xv[8], s = 0.f, s2 = 0.f;
#pragma unroll
    for (int t = 0; t < 8; ++t) {
      int d = lane + t * 32;
      float x = sBuf[wave][rr][d] + parent[(size_t)g * DIM + d] + bout[d];
      xv[t] = x; s += x; s2 += x * x;
    }
#pragma unroll
    for (int off = 16; off >= 1; off >>= 1) {
      s  += __shfl_xor(s,  off, 32);
      s2 += __shfl_xor(s2, off, 32);
    }
    const float mean = s * (1.0f / DIM);
    const float var  = s2 * (1.0f / DIM) - mean * mean;
    const float rstd = rsqrtf(var + 1e-5f);
#pragma unroll
    for (int t = 0; t < 8; ++t) {
      int d = lane + t * 32;
      out[(size_t)g * DIM + d] = (xv[t] - mean) * rstd * gamma[d] + beta[d];
    }
  }
}

// ---------------------------------------------------------------------------
extern "C" void kernel_launch(void* const* d_in, const int* in_sizes, int n_in,
                              void* d_out, int out_size, void* d_ws, size_t ws_size,
                              hipStream_t stream) {
  const float*     parent = (const float*)d_in[0];
  const float*     child  = (const float*)d_in[1];
  const long long* cidx   = (const long long*)d_in[2];   // int64 per reference
  const float*     Wq     = (const float*)d_in[3];
  const float*     Wkv    = (const float*)d_in[4];
  const float*     pos    = (const float*)d_in[5];
  const float*     Wout   = (const float*)d_in[6];
  const float*     boutp  = (const float*)d_in[7];
  const float*     gamma  = (const float*)d_in[8];
  const float*     beta   = (const float*)d_in[9];
  float* out = (float*)d_out;

  const int N = in_sizes[0] / DIM;                 // 16384
  const size_t WSZ = (size_t)DIM * DIM;            // 65536 weights each

  // Workspace: ctx [N*DIM f32], then bf16 weights: Wkv v-half, Wq, Wout.
  // q scratch lives in d_out (consumed by attn before outproj overwrites it).
  char*   ws     = (char*)d_ws;
  float*  ctx_ws = (float*)ws;
  __bf16* wkvv   = (__bf16*)(ws + (size_t)N * DIM * sizeof(float));
  __bf16* wqb    = wkvv + WSZ;
  __bf16* woutb  = wqb + WSZ;
  float*  q_ws   = out;

  cvt_weights<<<(int)(WSZ / 256), 256, 0, stream>>>(Wkv, Wq, Wout,
                                                    wkvv, wqb, woutb);
  qproj_wmma<<<N / 128, 256, 0, stream>>>(parent, wqb, q_ws);
  attn_wmma<<<N / RB, 256, 0, stream>>>(child, cidx, pos, Wkv, q_ws, wkvv, ctx_ws);
  outproj_ln<<<N / 128, 256, 0, stream>>>(ctx_ws, woutb, boutp, parent,
                                          gamma, beta, out);
}
